// MultiQueryAttention_19026705122065
// MI455X (gfx1250) — matrix operand order repaired and checked by reading, not compiled, not run on hardware
//
#include <hip/hip_runtime.h>
#include <stdint.h>
#include <stddef.h>

// ---------------------------------------------------------------------------
// MQA (B=2, S=2048, D=2048, H=16, HD=128) for MI455X / gfx1250, wave32.
// ~142 GFLOP vs ~150MB traffic -> matrix-unit bound; everything runs through
// v_wmma_f32_16x16x32_bf16 (fp32 accumulate).
//
// GEMM: block-cooperative. 8 waves share a 64-col weight panel staged in LDS
// via GLOBAL_LOAD_ASYNC_TO_LDS_B128 (ASYNCcnt, double-buffered, s_wait_asynccnt
// + block barriers). Each wave computes a 32x64 C tile (2x4 v8f accumulators),
// A fragments streamed from L2 (weights <=8MB bf16 sit in the 192MB L2).
//
// Attention: flash-style, one wave per (b,h,16 q rows); P relayout via LDS.
// ---------------------------------------------------------------------------

#define B_  2
#define S_  2048
#define D_  2048
#define H_  16
#define HD_ 128

#define USE_ASYNC_LDS 1   // set 0 to fall back to sync LDS fill

typedef __bf16 bf16;
typedef __attribute__((ext_vector_type(16))) __bf16 v16bf;
typedef __attribute__((ext_vector_type(8)))  __bf16 v8bf;
typedef __attribute__((ext_vector_type(8)))  float  v8f;

union V16 { v16bf v; v8bf h[2]; };

static __device__ __forceinline__ bf16 f2bf(float f) {
  union { float f; unsigned int u; } a{f};
  unsigned int r = (a.u + 0x7FFFu + ((a.u >> 16) & 1u)) >> 16;  // RNE
  union { unsigned short s; bf16 b; } o;
  o.s = (unsigned short)r;
  return o.b;
}

static __device__ __forceinline__ v8f wmma_bf16(v16bf a, v16bf b, v8f c) {
  // D = A(16x32) * B(32x16) + C, fp32 accum
  return __builtin_amdgcn_wmma_f32_16x16x32_bf16(
      /*neg_a=*/false, a, /*neg_b=*/false, b,
      /*c_mod=*/(short)0, c, /*reuse_a=*/false, /*reuse_b=*/false);
}

// Async global->LDS 16B copy (per-lane). LDS dest VGPR = low 32 bits of the
// generic pointer (ISA 10.2: LDS aperture decodes addr[31:0] as LDS offset).
static __device__ __forceinline__ void async_copy_b128(const void* gsrc, void* lds_dst) {
#if USE_ASYNC_LDS
  unsigned lds_off = (unsigned)(uintptr_t)lds_dst;
  asm volatile("global_load_async_to_lds_b128 %0, %1, off"
               :: "v"(lds_off), "v"(gsrc) : "memory");
#else
  *(uint4*)lds_dst = *(const uint4*)gsrc;
#endif
}

#if USE_ASYNC_LDS
#define WAIT_ASYNC(n) asm volatile("s_wait_asynccnt %0" :: "i"(n) : "memory")
#else
#define WAIT_ASYNC(n) ((void)0)
#endif

// ----------------------------- elementwise casts ---------------------------

__global__ void cast_bf16_kernel(const float* __restrict__ in,
                                 bf16* __restrict__ out, int n) {
  int i = blockIdx.x * blockDim.x + threadIdx.x;
  if (i < n) out[i] = f2bf(in[i]);
}

// W is (K x N) row-major fp32 -> WT (N x K) row-major bf16
__global__ void transpose_cast_kernel(const float* __restrict__ W,
                                      bf16* __restrict__ WT, int K, int N) {
  int i = blockIdx.x * blockDim.x + threadIdx.x;
  if (i < K * N) {
    int k = i / N, n = i - k * N;
    WT[(size_t)n * K + k] = f2bf(W[i]);
  }
}

// ------------------------------- WMMA GEMM ---------------------------------
// C(MxN) = A(MxK,bf16) @ BT^T (BT is NxK bf16) + bias(N)
// Block: 256 threads = 8 waves, computes 256 rows x 64 cols.
// Weight panel (64 cols x 32 K) staged in LDS, double-buffered, async-filled.
// STORE_T: store transposed as [b][N][Srows] (for V^T). OUT_F32: fp32 output.

template <bool STORE_T, bool OUT_F32>
__global__ void __launch_bounds__(256)
gemm_bf16_wmma(const bf16* __restrict__ A, const bf16* __restrict__ BT,
               const float* __restrict__ bias, void* __restrict__ Cout,
               int M, int N, int K, int Srows) {
  __shared__ __align__(16) bf16 Bsh[2][64][32];  // 2 x 4KB panels

  const int tid  = threadIdx.x;
  const int lane = tid & 31;
  const int ln   = lane & 15;   // A: row in tile / B,C: column in tile
  const int hi   = lane >> 4;   // half-lane selects K sub-range / C row half
  const int w    = tid >> 5;

  const int ntilesN = N >> 6;
  const int bm = blockIdx.x / ntilesN;
  const int bn = blockIdx.x - bm * ntilesN;
  const int n0 = bn << 6;
  const int m0 = (bm << 8) + (w << 5);  // this wave's 32 rows

  // cooperative fill mapping: thread -> (column, 16B segment of its 32-K row)
  const int fc = tid >> 2;
  const int fs = (tid & 3) * 8;
  const bf16* fill_base = BT + (size_t)(n0 + fc) * K + fs;

  v8f acc[2][4] = {};
  const bf16* arow0 = A + (size_t)(m0 + ln) * K;
  const bf16* arow1 = A + (size_t)(m0 + 16 + ln) * K;

  // prologue: stage K-chunk 0 into buffer 0
  async_copy_b128(fill_base, &Bsh[0][fc][fs]);

  for (int kk = 0; kk < K; kk += 32) {
    const int cur = (kk >> 5) & 1;
    if (kk + 32 < K) {
      async_copy_b128(fill_base + kk + 32, &Bsh[1 - cur][fc][fs]);
      WAIT_ASYNC(1);  // in-order completion: current panel is resident
    } else {
      WAIT_ASYNC(0);
    }
    __syncthreads();  // panel visible to (and awaited by) all 8 waves

    // A fragments (16x32): lane holds row ln, K = kk + hi*16 + {0..15}
    V16 a0, a1;
    a0.h[0] = *(const v8bf*)(arow0 + kk + hi * 16);
    a0.h[1] = *(const v8bf*)(arow0 + kk + hi * 16 + 8);
    a1.h[0] = *(const v8bf*)(arow1 + kk + hi * 16);
    a1.h[1] = *(const v8bf*)(arow1 + kk + hi * 16 + 8);
    if (kk + 64 <= K) __builtin_prefetch(arow0 + kk + 64, 0, 1);  // global_prefetch_b8

#pragma unroll
    for (int s = 0; s < 4; ++s) {
      // B fragment (32x16) from LDS: lane holds col s*16+ln, K = hi*16+{0..15}
      V16 b;
      b.h[0] = *(const v8bf*)&Bsh[cur][s * 16 + ln][hi * 16];
      b.h[1] = *(const v8bf*)&Bsh[cur][s * 16 + ln][hi * 16 + 8];
      acc[0][s] = wmma_bf16(a0.v, b.v, acc[0][s]);
      acc[1][s] = wmma_bf16(a1.v, b.v, acc[1][s]);
    }
    __syncthreads();  // all waves done reading 'cur' before it is re-staged
  }

#pragma unroll
  for (int rt = 0; rt < 2; ++rt) {
#pragma unroll
    for (int s = 0; s < 4; ++s) {
      const int c = n0 + s * 16 + ln;
      const float bval = bias ? bias[c] : 0.0f;
#pragma unroll
      for (int j = 0; j < 8; ++j) {
        const int r = m0 + rt * 16 + hi * 8 + j;  // C layout: vgpr j -> rows j/j+8
        float val = acc[rt][s][j] + bval;
        if constexpr (OUT_F32) {
          ((float*)Cout)[(size_t)r * N + c] = val;
        } else if constexpr (STORE_T) {
          const int bb = r / Srows, ss = r - (r / Srows) * Srows;
          ((bf16*)Cout)[(size_t)bb * N * Srows + (size_t)c * Srows + ss] = f2bf(val);
        } else {
          ((bf16*)Cout)[(size_t)r * N + c] = f2bf(val);
        }
      }
    }
  }
}

// ------------------------------ MQA attention ------------------------------
// One wave per (b, h, 16 q rows). Stream keys 32 at a time:
//   S = Q K^T * scale  (8 WMMA over HD=128) -> exp -> P staged via LDS ->
//   O += P V           (8 WMMA over the 8 d-tiles of HD)
// Running per-row sum (scores are O(1) by construction -> exp cannot
// overflow; mask handled as exact zero).

__global__ void __launch_bounds__(256)
mqa_attn_wmma(const bf16* __restrict__ qh,   // [B,S,D]  (d = h*HD + hd)
              const bf16* __restrict__ kh,   // [B,S,HD]
              const bf16* __restrict__ vT,   // [B,HD,S]
              const uint8_t* __restrict__ mask,  // [B,S]
              bf16* __restrict__ Oh) {       // [B,S,D]
  __shared__ __align__(16) bf16 Pl[8][16][32];  // per-wave P tile (8KB total)

  const int lane = threadIdx.x & 31;
  const int ln   = lane & 15;
  const int hi   = lane >> 4;
  const int w    = threadIdx.x >> 5;
  const int wid  = blockIdx.x * 8 + w;
  const int qt   = wid & ((S_ / 16) - 1);
  const int bh   = wid / (S_ / 16);
  const int h    = bh & (H_ - 1);
  const int b    = bh / H_;
  const int q0   = qt << 4;

  // Q A-fragments: 16 rows x HD=128, as 4 chunks of K=32. Loaded once.
  V16 aq[4];
  const bf16* qbase = qh + ((size_t)(b * S_ + q0 + ln)) * D_ + h * HD_;
#pragma unroll
  for (int c = 0; c < 4; ++c) {
    aq[c].h[0] = *(const v8bf*)(qbase + c * 32 + hi * 16);
    aq[c].h[1] = *(const v8bf*)(qbase + c * 32 + hi * 16 + 8);
  }

  v8f oacc[8] = {};                       // 16 x 128 fp32 accumulators
  float lsum[8] = {0, 0, 0, 0, 0, 0, 0, 0};
  const float scale = 0.08838834764831845f;  // 1/sqrt(128)

  for (int kt = 0; kt < S_; kt += 32) {
#pragma unroll
    for (int sub = 0; sub < 2; ++sub) {
      const int kc0 = kt + sub * 16;
      v8f sacc = {};
#pragma unroll
      for (int c = 0; c < 4; ++c) {
        // B frag: lane col = key kc0+ln, K = d in [c*32 + hi*16, +16) contiguous
        const bf16* kp = kh + ((size_t)(b * S_ + kc0 + ln)) * HD_ + c * 32 + hi * 16;
        V16 bk;
        bk.h[0] = *(const v8bf*)kp;
        bk.h[1] = *(const v8bf*)(kp + 8);
        sacc = wmma_bf16(aq[c].v, bk.v, sacc);
      }
      // This lane's score column = key kc0+ln for all 8 accum rows.
      const float mcol = mask[b * S_ + kc0 + ln] ? 1.0f : 0.0f;
#pragma unroll
      for (int j = 0; j < 8; ++j) {
        const float p = mcol * __expf(sacc[j] * scale);  // v_exp_f32
        lsum[j] += p;                                    // partial row sum
        Pl[w][hi * 8 + j][sub * 16 + ln] = f2bf(p);      // C-layout -> LDS
      }
    }
    // Same-wave DS ordering is in-order; fence keeps compiler from reordering
    // and guarantees staged P is visible to the fragment reloads below.
    asm volatile("s_wait_dscnt 0" ::: "memory");

    // Reload P in A-fragment layout: row = ln, K = hi*16 + {0..15}
    V16 ap;
    ap.h[0] = *(const v8bf*)&Pl[w][ln][hi * 16];
    ap.h[1] = *(const v8bf*)&Pl[w][ln][hi * 16 + 8];
#pragma unroll
    for (int nt = 0; nt < 8; ++nt) {
      // B frag from V^T: lane col = d = nt*16+ln, K = keys kt + hi*16 + {0..15}
      const bf16* vp = vT + (size_t)b * HD_ * S_ + (size_t)(nt * 16 + ln) * S_
                       + kt + hi * 16;
      V16 bv;
      bv.h[0] = *(const v8bf*)vp;
      bv.h[1] = *(const v8bf*)(vp + 8);
      oacc[nt] = wmma_bf16(ap.v, bv.v, oacc[nt]);
    }
  }

  // Full row sums: reduce partials across each 16-lane group.
#pragma unroll
  for (int j = 0; j < 8; ++j) {
    float s = lsum[j];
    s += __shfl_xor(s, 1, 16);
    s += __shfl_xor(s, 2, 16);
    s += __shfl_xor(s, 4, 16);
    s += __shfl_xor(s, 8, 16);
    lsum[j] = s > 0.0f ? 1.0f / s : 0.0f;
  }

#pragma unroll
  for (int nt = 0; nt < 8; ++nt) {
    const int d = h * HD_ + nt * 16 + ln;
#pragma unroll
    for (int j = 0; j < 8; ++j) {
      const int qr = q0 + hi * 8 + j;
      Oh[((size_t)(b * S_ + qr)) * D_ + d] = f2bf(oacc[nt][j] * lsum[j]);
    }
  }
}

// ------------------------------- launcher ----------------------------------

extern "C" void kernel_launch(void* const* d_in, const int* in_sizes, int n_in,
                              void* d_out, int out_size, void* d_ws, size_t ws_size,
                              hipStream_t stream) {
  const float*   x    = (const float*)d_in[0];
  const uint8_t* mask = (const uint8_t*)d_in[1];
  const float*   Wq   = (const float*)d_in[2];
  const float*   bq   = (const float*)d_in[3];
  const float*   Wk   = (const float*)d_in[4];
  const float*   bk   = (const float*)d_in[5];
  const float*   Wv   = (const float*)d_in[6];
  const float*   bv   = (const float*)d_in[7];
  const float*   Wo   = (const float*)d_in[8];
  const float*   bo   = (const float*)d_in[9];
  (void)in_sizes; (void)n_in; (void)out_size; (void)ws_size;

  // Workspace carve (~71 MB): bf16 copies of activations/weights + intermediates.
  char* p = (char*)d_ws;
  auto carve = [&](size_t elems) {
    bf16* r = (bf16*)p;
    p += ((elems * sizeof(bf16)) + 255) & ~(size_t)255;
    return r;
  };
  bf16* xh  = carve((size_t)B_ * S_ * D_);
  bf16* WqT = carve((size_t)D_ * D_);
  bf16* WkT = carve((size_t)D_ * HD_);
  bf16* WvT = carve((size_t)D_ * HD_);
  bf16* WoT = carve((size_t)D_ * D_);
  bf16* qhb = carve((size_t)B_ * S_ * D_);
  bf16* khb = carve((size_t)B_ * S_ * HD_);
  bf16* vTb = carve((size_t)B_ * HD_ * S_);
  bf16* Oh  = carve((size_t)B_ * S_ * D_);

  const int M = B_ * S_;  // 4096 rows

  { int n = B_ * S_ * D_;
    cast_bf16_kernel<<<(n + 255) / 256, 256, 0, stream>>>(x, xh, n); }
  transpose_cast_kernel<<<(D_ * D_  + 255) / 256, 256, 0, stream>>>(Wq, WqT, D_, D_);
  transpose_cast_kernel<<<(D_ * HD_ + 255) / 256, 256, 0, stream>>>(Wk, WkT, D_, HD_);
  transpose_cast_kernel<<<(D_ * HD_ + 255) / 256, 256, 0, stream>>>(Wv, WvT, D_, HD_);
  transpose_cast_kernel<<<(D_ * D_  + 255) / 256, 256, 0, stream>>>(Wo, WoT, D_, D_);

  // Q projection: (M x D) = xh @ Wq + bq, bf16 out, [B,S,D]
  { int blocks = (M / 256) * (D_ / 64);
    gemm_bf16_wmma<false, false><<<blocks, 256, 0, stream>>>(
        xh, WqT, bq, qhb, M, D_, D_, S_); }
  // K projection: (M x HD), bf16 out, [B,S,HD]
  { int blocks = (M / 256) * (HD_ / 64);
    gemm_bf16_wmma<false, false><<<blocks, 256, 0, stream>>>(
        xh, WkT, bk, khb, M, HD_, D_, S_); }
  // V projection stored transposed: [B,HD,S]
  { int blocks = (M / 256) * (HD_ / 64);
    gemm_bf16_wmma<true, false><<<blocks, 256, 0, stream>>>(
        xh, WvT, bv, vTb, M, HD_, D_, S_); }

  // Attention: one wave per (b, h, 16-row q tile)
  { int tiles = B_ * H_ * (S_ / 16);
    mqa_attn_wmma<<<tiles / 8, 256, 0, stream>>>(qhb, khb, vTb, mask, Oh); }

  // Output projection -> fp32 d_out
  { int blocks = (M / 256) * (D_ / 64);
    gemm_bf16_wmma<false, true><<<blocks, 256, 0, stream>>>(
        Oh, WoT, bo, d_out, M, D_, D_, S_); }
}
